// SparseConv3d_19318762897756
// MI455X (gfx1250) — matrix operand order repaired and checked by reading, not compiled, not run on hardware
//
#include <hip/hip_runtime.h>

// ---------------------------------------------------------------------------
// CDNA5 (gfx1250, wave32) sparse 3D conv chain + BN + ReLU
//   h = conv_z(feats,W1); h = conv_y(h,W2); h = conv_x(h,W3); BN+ReLU
// Matmuls: v_wmma_f32_16x16x32_bf16 with bf16 hi/lo split of both operands
// (3 WMMAs per K-chunk -> ~fp32 accuracy at memory-roofline speed).
// Weights are pre-split/pre-swizzled once into a global fragment buffer,
// then DMA'd into LDS per block with the Tensor Data Mover (TDM).
// ---------------------------------------------------------------------------

typedef __attribute__((ext_vector_type(16))) __bf16       v16bf;
typedef __attribute__((ext_vector_type(8)))  float        v8f;
typedef __attribute__((ext_vector_type(4)))  unsigned int u32x4;
typedef __attribute__((ext_vector_type(8)))  int          i32x8;
typedef __attribute__((ext_vector_type(4)))  int          i32x4;

#define CIN   64
#define COUT  64
#define WAVES 8          // 256 threads / block
#define TILE  16         // points per wave

// Weight fragment store: [tap(3)][chunk(2)][ctile(4)][lane(32)][elem(16)] bf16
#define WFRAG_ELEMS (3 * 2 * 4 * 32 * 16)   // == 3*64*64 = 12288 per hi/lo
#define WFRAG_BYTES (2 * WFRAG_ELEMS * 2)   // hi+lo = 49152 B = 12288 dwords
#define WFRAG_DW    (WFRAG_BYTES / 4)       // 12288

__device__ __forceinline__ float bf16_hi(float x, __bf16* h) {
    __bf16 b = (__bf16)x;           // RNE convert
    *h = b;
    return x - (float)b;            // residual (exact in fp32)
}

// ---------------------------------------------------------------------------
// One-time weight preprocessing: split each W [3,64,64] into bf16 hi/lo and
// lay it out in exact WMMA-B fragment order so conv blocks can bulk-copy it.
// blockIdx.x selects the layer.
// ---------------------------------------------------------------------------
__global__ __launch_bounds__(256)
void weight_prep_kernel(const float* __restrict__ W1,
                        const float* __restrict__ W2,
                        const float* __restrict__ W3,
                        __bf16* __restrict__ frag)
{
    const float* W = (blockIdx.x == 0) ? W1 : (blockIdx.x == 1) ? W2 : W3;
    __bf16* fhi = frag + (size_t)blockIdx.x * 2 * WFRAG_ELEMS;
    __bf16* flo = fhi + WFRAG_ELEMS;

    // B operand of 16x16x32: 32(K=cin) x 16(N=cout). Lane l<16 holds column
    // N=l with K = chunk*32 + e (e=0..15); lane l+16 holds K = chunk*32+16+e.
    for (int j = threadIdx.x; j < WFRAG_ELEMS; j += 256) {
        int e     =  j        & 15;
        int lane  = (j >> 4)  & 31;
        int ct    = (j >> 9)  & 3;
        int chunk = (j >> 11) & 1;
        int tap   =  j >> 12;
        int half  = lane >> 4;
        int n     = ct * 16 + (lane & 15);            // cout
        int k     = chunk * 32 + half * 16 + e;       // cin
        float w   = W[(tap * CIN + k) * COUT + n];
        __bf16 h;
        float  r  = bf16_hi(w, &h);
        fhi[j] = h;
        flo[j] = (__bf16)r;
    }
}

__global__ __launch_bounds__(256)
void conv_tap3_kernel(const float*  __restrict__ src,   // [N,64] fp32
                      const __bf16* __restrict__ frag,  // [2*WFRAG_ELEMS] hi|lo
                      const int*    __restrict__ nbr,   // [N,3] (-1 = missing)
                      float*        __restrict__ dst,   // [N,64] fp32
                      int ntiles)
{
    __shared__ __attribute__((aligned(128))) __bf16 sW[2 * WFRAG_ELEMS];
    const __bf16* sWhi = sW;
    const __bf16* sWlo = sW + WFRAG_ELEMS;

#if defined(__has_builtin) && __has_builtin(__builtin_amdgcn_tensor_load_to_lds)
    // ---- TDM: DMA the 48 KB pre-swizzled fragment blob into LDS ----------
    if (threadIdx.x < 32) {                 // wave 0 issues the tensor DMA
        unsigned long long ga = (unsigned long long)(size_t)(const void*)frag;
        unsigned int lds_addr = (unsigned int)(size_t)(void*)&sW[0];
        // D# group0: count=1 | lds_addr | global_addr[56:0] | type=2
        u32x4 g0 = { 1u,
                     lds_addr,
                     (unsigned int)(ga & 0xffffffffu),
                     ((unsigned int)((ga >> 32) & 0x01ffffffu)) | (2u << 30) };
        // D# group1: data_size=4B; tensor_dim0=tile_dim0=12288 dwords; dim1=1
        i32x8 g1 = { (int)(2u << 16),              // workgroup_mask=0, 4B elems
                     (int)((unsigned)WFRAG_DW << 16),   // tensor_dim0[15:0]
                     (int)(1u << 16),              // tensor_dim1 = 1
                     (int)((unsigned)WFRAG_DW << 16),   // tile_dim0
                     1,                            // tile_dim1 = 1
                     WFRAG_DW,                     // tensor_dim0_stride
                     0, 0 };
        i32x4 z4 = { 0, 0, 0, 0 };
        i32x8 z8 = { 0, 0, 0, 0, 0, 0, 0, 0 };
        __builtin_amdgcn_tensor_load_to_lds(g0, g1, z4, z4, z8, 0);
        __builtin_amdgcn_s_wait_tensorcnt(0);
    }
#else
    for (int j = threadIdx.x * 8; j < 2 * WFRAG_ELEMS; j += 256 * 8)
        *(uint4*)(sW + j) = *(const uint4*)(frag + j);
#endif
    __syncthreads();

    const int tid    = threadIdx.x;
    const int waveId = tid >> 5;
    const int lane   = tid & 31;
    const int half   = lane >> 4;        // 0: K-low half rows, 1: K-high
    const int m      = lane & 15;        // point row within tile / cout col

    const int t = blockIdx.x * WAVES + waveId;   // uniform per wave
    if (t >= ntiles) return;

    v8f acc0 = {}, acc1 = {}, acc2 = {}, acc3 = {};

    #pragma unroll
    for (int tap = 0; tap < 3; ++tap) {
        // Gather row for point m at this tap (lane L and L+16 share a row)
        const int row   = nbr[(t * TILE + m) * 3 + tap];
        const float zm  = (row >= 0) ? 1.0f : 0.0f;
        const float* rp = src + (size_t)((row >= 0) ? row : 0) * CIN;

        #pragma unroll
        for (int chunk = 0; chunk < 2; ++chunk) {
            // ---- Build A fragment (16 points x 32 cin), bf16 hi + lo ----
            // A layout: lane m holds row M=m; VGPR v holds the aligned K pair
            //   K = chunk*32 + 2v + 16*half  (+0,+1)
            v16bf ahi, alo;
            #pragma unroll
            for (int v = 0; v < 8; ++v) {
                int kk = chunk * 32 + 2 * v + 16 * half;
                float2 w = *(const float2*)(rp + kk);   // 8B-aligned pair
                float x0 = w.x * zm, x1 = w.y * zm;
                __bf16 h0, h1;
                float r0 = bf16_hi(x0, &h0);
                float r1 = bf16_hi(x1, &h1);
                ahi[2 * v]     = h0;  alo[2 * v]     = (__bf16)r0;
                ahi[2 * v + 1] = h1;  alo[2 * v + 1] = (__bf16)r1;
            }

            // ---- 4 cout tiles x 3 split-WMMAs each ----
            const int fb = ((tap * 2 + chunk) * 4) * 32 * 16 + lane * 16;
            #pragma unroll
            for (int ct = 0; ct < 4; ++ct) {
                const v16bf bhi = *(const v16bf*)(sWhi + fb + ct * 32 * 16);
                const v16bf blo = *(const v16bf*)(sWlo + fb + ct * 32 * 16);
                v8f* accp = (ct == 0) ? &acc0 : (ct == 1) ? &acc1
                          : (ct == 2) ? &acc2 : &acc3;
                v8f c = *accp;
                c = __builtin_amdgcn_wmma_f32_16x16x32_bf16(
                        false, ahi, false, bhi, (short)0, c, false, false);
                c = __builtin_amdgcn_wmma_f32_16x16x32_bf16(
                        false, ahi, false, blo, (short)0, c, false, false);
                c = __builtin_amdgcn_wmma_f32_16x16x32_bf16(
                        false, alo, false, bhi, (short)0, c, false, false);
                *accp = c;
            }
        }
    }

    // ---- Store: D layout — lane (half,m), VGPR v -> (M = v + 8*half, N = m)
    #pragma unroll
    for (int ct = 0; ct < 4; ++ct) {
        const v8f c = (ct == 0) ? acc0 : (ct == 1) ? acc1
                    : (ct == 2) ? acc2 : acc3;
        #pragma unroll
        for (int v = 0; v < 8; ++v) {
            int prow = t * TILE + v + 8 * half;
            dst[(size_t)prow * COUT + ct * 16 + m] = c[v];
        }
    }
}

// ---------------------------------------------------------------------------
// BatchNorm statistics + normalize/ReLU
// ---------------------------------------------------------------------------
__global__ void stats_init_kernel(float* stats)
{
    if (threadIdx.x < 2 * COUT) stats[threadIdx.x] = 0.0f;
}

__global__ __launch_bounds__(256)
void stats_kernel(const float* __restrict__ h, float* __restrict__ stats, int N)
{
    const int c  = threadIdx.x & 63;
    const int rs = blockIdx.x * 4 + (threadIdx.x >> 6);
    const int stride = gridDim.x * 4;
    float s = 0.0f, q = 0.0f;
    for (int r = rs; r < N; r += stride) {
        float x = h[(size_t)r * COUT + c];
        s += x;
        q += x * x;
    }
    __shared__ float ls[256], lq[256];
    ls[threadIdx.x] = s;
    lq[threadIdx.x] = q;
    __syncthreads();
    if (threadIdx.x < 64) {
        s = ls[threadIdx.x] + ls[threadIdx.x + 64] +
            ls[threadIdx.x + 128] + ls[threadIdx.x + 192];
        q = lq[threadIdx.x] + lq[threadIdx.x + 64] +
            lq[threadIdx.x + 128] + lq[threadIdx.x + 192];
        atomicAdd(&stats[c], s);
        atomicAdd(&stats[COUT + c], q);
    }
}

__global__ __launch_bounds__(256)
void bn_relu_kernel(const float* __restrict__ h,
                    const float* __restrict__ stats,
                    const float* __restrict__ gamma,
                    const float* __restrict__ beta,
                    float* __restrict__ out, int N)
{
    const float invN = 1.0f / (float)N;
    size_t total = (size_t)N * COUT;
    for (size_t i = (size_t)blockIdx.x * 256 + threadIdx.x; i < total;
         i += (size_t)gridDim.x * 256) {
        int   c    = (int)(i & 63);
        float mean = stats[c] * invN;
        float var  = stats[COUT + c] * invN - mean * mean;
        float sc   = gamma[c] * rsqrtf(var + 1e-5f);
        float x    = (h[i] - mean) * sc + beta[c];
        out[i] = fmaxf(x, 0.0f);
    }
}

// ---------------------------------------------------------------------------
extern "C" void kernel_launch(void* const* d_in, const int* in_sizes, int n_in,
                              void* d_out, int out_size, void* d_ws, size_t ws_size,
                              hipStream_t stream)
{
    const float* feats = (const float*)d_in[0];
    const float* W1    = (const float*)d_in[1];
    const float* W2    = (const float*)d_in[2];
    const float* W3    = (const float*)d_in[3];
    const float* gamma = (const float*)d_in[4];
    const float* beta  = (const float*)d_in[5];
    const int*   nbr_z = (const int*)d_in[6];
    const int*   nbr_y = (const int*)d_in[7];
    const int*   nbr_x = (const int*)d_in[8];

    const int N      = in_sizes[0] / CIN;   // 1,000,000 (divisible by 16)
    const int ntiles = N / TILE;

    float*  h_ws  = (float*)d_ws;                    // [N,64] fp32 scratch
    float*  stats = h_ws + (size_t)N * COUT;         // [128] sum / sumsq
    __bf16* frag  = (__bf16*)(stats + 2 * COUT);     // 3 x 48KB fragment blobs
    float*  outp  = (float*)d_out;

    const int blocks = (ntiles + WAVES - 1) / WAVES;

    // one-time weight split into WMMA fragment layout (3 layers)
    weight_prep_kernel<<<3, 256, 0, stream>>>(W1, W2, W3, frag);

    // conv chain: feats -> ws -> d_out -> ws   (ping-pong, d_out as scratch)
    conv_tap3_kernel<<<blocks, 256, 0, stream>>>(feats, frag,
                                                 nbr_z, h_ws, ntiles);
    conv_tap3_kernel<<<blocks, 256, 0, stream>>>(h_ws, frag + 2 * WFRAG_ELEMS,
                                                 nbr_y, outp, ntiles);
    conv_tap3_kernel<<<blocks, 256, 0, stream>>>(outp, frag + 4 * WFRAG_ELEMS,
                                                 nbr_x, h_ws, ntiles);

    // batch-norm statistics over N rows, then normalize + relu into d_out
    stats_init_kernel<<<1, 128, 0, stream>>>(stats);
    stats_kernel<<<1024, 256, 0, stream>>>(h_ws, stats, N);
    bn_relu_kernel<<<2048, 256, 0, stream>>>(h_ws, stats, gamma, beta, outp, N);
}
